// MyGNNLayer_82377472738077
// MI455X (gfx1250) — matrix operand order repaired and checked by reading, not compiled, not run on hardware
//
#include <hip/hip_runtime.h>
#include <math.h>

typedef __attribute__((ext_vector_type(16))) __bf16 v16bf;
typedef __attribute__((ext_vector_type(8)))  float  v8f;

#define Hdim 64
#define NN 100000
#define NE 1000000
#define NG 64
#define EDGE_TILES (NE/16)
#define NODE_TILES (NN/16)

struct Acc { v8f c[4]; };

__device__ inline void acc_zero(Acc& a) {
  v8f z = {0.f,0.f,0.f,0.f,0.f,0.f,0.f,0.f};
  a.c[0]=z; a.c[1]=z; a.c[2]=z; a.c[3]=z;
}

__device__ inline void cvt8(const float* __restrict__ s, __bf16* d) {
  float4 a = ((const float4*)s)[0];
  float4 b = ((const float4*)s)[1];
  d[0]=(__bf16)a.x; d[1]=(__bf16)a.y; d[2]=(__bf16)a.z; d[3]=(__bf16)a.w;
  d[4]=(__bf16)b.x; d[5]=(__bf16)b.y; d[6]=(__bf16)b.z; d[7]=(__bf16)b.w;
}
__device__ inline void cvt8s(const float* __restrict__ s, float sc, __bf16* d) {
  float4 a = ((const float4*)s)[0];
  float4 b = ((const float4*)s)[1];
  d[0]=(__bf16)(a.x*sc); d[1]=(__bf16)(a.y*sc); d[2]=(__bf16)(a.z*sc); d[3]=(__bf16)(a.w*sc);
  d[4]=(__bf16)(b.x*sc); d[5]=(__bf16)(b.y*sc); d[6]=(__bf16)(b.z*sc); d[7]=(__bf16)(b.w*sc);
}

union AF { v16bf v; __bf16 h[16]; };
union BF { v16bf v; float4 f[2]; };

// A fragment (16x32 bf16, MxK): lane = M&15; lane-half selects K chunks
// {kt*32+khalf .. +8} and {kt*32+16+khalf .. +8}, khalf = 8*(lane>>4).
__device__ inline v16bf a_frag_segs(const float* const* segs, int kt, int lane) {
  AF af;
  int khalf = (lane >> 4) << 3;
  int k0 = kt*32 + khalf;
  int k1 = k0 + 16;
  cvt8(segs[k0 >> 6] + (k0 & 63), &af.h[0]);
  cvt8(segs[k1 >> 6] + (k1 & 63), &af.h[8]);
  return af.v;
}
__device__ inline v16bf a_frag_seg(const float* seg, int ktLocal, int lane) {
  AF af;
  int khalf = (lane >> 4) << 3;
  int k0 = ktLocal*32 + khalf;
  cvt8(seg + k0,      &af.h[0]);
  cvt8(seg + k0 + 16, &af.h[8]);
  return af.v;
}
__device__ inline v16bf a_frag_seg_scaled(const float* seg, float sc, int ktLocal, int lane) {
  AF af;
  int khalf = (lane >> 4) << 3;
  int k0 = ktLocal*32 + khalf;
  cvt8s(seg + k0,      sc, &af.h[0]);
  cvt8s(seg + k0 + 16, sc, &af.h[8]);
  return af.v;
}
__device__ inline v16bf a_frag_lds(const __bf16* hs, int ktLocal, int lane) {
  int m = lane & 15;
  int khalf = (lane >> 4) << 3;
  int k0 = ktLocal*32 + khalf;
  BF bf;
  bf.f[0] = *(const float4*)(hs + m*Hdim + k0);
  bf.f[1] = *(const float4*)(hs + m*Hdim + k0 + 16);
  return bf.v;
}
// B fragment (32x16 bf16, KxN): lane = N&15; lane-half selects K chunks
// {khalf .. +8} and {16+khalf .. +8}, khalf = 8*(lane>>4); Wt is N-major (transposed weight).
__device__ inline v16bf b_frag(const __bf16* Wt, int ldk, int nt, int kt, int lane) {
  const __bf16* p = Wt + (size_t)(nt*16 + (lane & 15))*ldk + kt*32 + ((lane >> 4) << 3);
  BF bf;
  bf.f[0] = ((const float4*)p)[0];
  bf.f[1] = ((const float4*)p)[2];
  return bf.v;
}

#define WMMA_BF16(A,B,C) __builtin_amdgcn_wmma_f32_16x16x32_bf16(false,(A),false,(B),(short)0,(C),false,false)

__device__ inline void bias_gelu(Acc& a, const float bn[4]) {
  #pragma unroll
  for (int nt = 0; nt < 4; ++nt) {
    #pragma unroll
    for (int v = 0; v < 8; ++v) {
      float xv = a.c[nt][v] + bn[nt];
      a.c[nt][v] = 0.5f * xv * (1.0f + erff(xv * 0.70710678118654752f));
    }
  }
}

__device__ inline void layernorm(Acc& a, const float gn[4], const float bn[4]) {
  #pragma unroll
  for (int v = 0; v < 8; ++v) {
    float s = a.c[0][v] + a.c[1][v] + a.c[2][v] + a.c[3][v];
    s += __shfl_xor(s, 1); s += __shfl_xor(s, 2); s += __shfl_xor(s, 4); s += __shfl_xor(s, 8);
    float mean = s * (1.0f / 64.0f);
    float q = 0.f;
    #pragma unroll
    for (int nt = 0; nt < 4; ++nt) { float d = a.c[nt][v] - mean; q += d*d; }
    q += __shfl_xor(q, 1); q += __shfl_xor(q, 2); q += __shfl_xor(q, 4); q += __shfl_xor(q, 8);
    float inv = rsqrtf(q * (1.0f / 64.0f) + 1e-5f);
    #pragma unroll
    for (int nt = 0; nt < 4; ++nt)
      a.c[nt][v] = (a.c[nt][v] - mean) * inv * gn[nt] + bn[nt];
  }
}

// C layout (lane = N&15, VGPR v = row v + 8*(lane>>4)) -> LDS [m][64] bf16.
__device__ inline void stage_bf16(__bf16* hs, const Acc& a, int lane) {
  int n0 = lane & 15;
  int mh = (lane >> 4) << 3;
  #pragma unroll
  for (int nt = 0; nt < 4; ++nt) {
    int n = n0 + 16*nt;
    #pragma unroll
    for (int v = 0; v < 8; ++v)
      hs[(mh + v)*Hdim + n] = (__bf16)a.c[nt][v];
  }
  asm volatile("s_wait_dscnt 0" ::: "memory");
}

__device__ inline void gemm64_lds(Acc& a, const __bf16* hs, const __bf16* Wt, int lane) {
  Acc o; acc_zero(o);
  #pragma unroll
  for (int kt = 0; kt < 2; ++kt) {
    v16bf av = a_frag_lds(hs, kt, lane);
    #pragma unroll
    for (int nt = 0; nt < 4; ++nt)
      o.c[nt] = WMMA_BF16(av, b_frag(Wt, Hdim, nt, kt, lane), o.c[nt]);
  }
  a = o;
}

__device__ inline void store_rows(float* __restrict__ out, int rowBase, const Acc& a, int lane) {
  int n0 = lane & 15;
  int mh = (lane >> 4) << 3;
  #pragma unroll
  for (int v = 0; v < 8; ++v) {
    #pragma unroll
    for (int nt = 0; nt < 4; ++nt)
      out[(size_t)(rowBase + mh + v)*Hdim + n0 + 16*nt] = a.c[nt][v];
  }
}

// cooperative global(bf16) -> LDS copy, 16B granules
__device__ inline void copy_lds(__bf16* dst, const __bf16* src, int n_bf16,
                                int tid, int nthreads) {
  int chunks = n_bf16 >> 3;
  for (int i = tid; i < chunks; i += nthreads)
    ((float4*)dst)[i] = ((const float4*)src)[i];
}

// load 4 per-lane coefficients (column n0+16*nt) into registers
__device__ inline void load_coef(float c[4], const float* __restrict__ p, int lane) {
  int n0 = lane & 15;
  #pragma unroll
  for (int nt = 0; nt < 4; ++nt) c[nt] = p[n0 + 16*nt];
}

// ---- prep: fp32 [K][64] weight -> bf16 N-major [64][K] ----
__global__ void cvt_transpose_kernel(const float* __restrict__ src,
                                     __bf16* __restrict__ dst, int K) {
  int i = blockIdx.x * blockDim.x + threadIdx.x;
  if (i < K * Hdim) {
    int k = i / Hdim, n = i % Hdim;
    dst[(size_t)n * K + k] = (__bf16)src[i];
  }
}

// ---- fused edge MLP + message MLP + scatter-add; persistent waves, 16 edges/tile ----
__global__ __launch_bounds__(128) void edge_node_msg_kernel(
    const float* __restrict__ x, const int* __restrict__ eidx,
    const float* __restrict__ eattr, const float* __restrict__ u,
    const int* __restrict__ batch,
    const __bf16* __restrict__ wAll,   // WE1|WE2|WM1|WM2 contiguous bf16 (32768 elems)
    const float* __restrict__ bE1, const float* __restrict__ bE2,
    const float* __restrict__ gE, const float* __restrict__ beE,
    const float* __restrict__ bM1, const float* __restrict__ bM2,
    const float* __restrict__ gM, const float* __restrict__ beM,
    float* __restrict__ edge_out, float* __restrict__ aggsum, float* __restrict__ cnt)
{
  __shared__ __bf16 smem[32768 + 4*16*Hdim];   // 64KB weights + 8KB staging
  const __bf16* sWE1 = smem;            // 16384 (256x64)
  const __bf16* sWE2 = smem + 16384;    //  4096 ( 64x64)
  const __bf16* sWM1 = smem + 20480;    //  8192 (128x64)
  const __bf16* sWM2 = smem + 28672;    //  4096 ( 64x64)
  int tid  = threadIdx.x;
  int lane = tid & 31;
  int wave = tid >> 5;
  __bf16* hs = smem + 32768 + wave * 16 * Hdim;

  copy_lds(smem, wAll, 32768, tid, 128);
  __syncthreads();

  int m0 = lane & 15;
  int mh = (lane >> 4) << 3;
  float cbE1[4], cbE2[4], cgE[4], cbeE[4], cbM1[4], cbM2[4], cgM[4], cbeM[4];
  load_coef(cbE1, bE1, lane); load_coef(cbE2, bE2, lane);
  load_coef(cgE,  gE,  lane); load_coef(cbeE, beE, lane);
  load_coef(cbM1, bM1, lane); load_coef(cbM2, bM2, lane);
  load_coef(cgM,  gM,  lane); load_coef(cbeM, beM, lane);

  int waveId = blockIdx.x * 4 + wave;
  int nWaves = gridDim.x * 4;
  for (int tile = waveId; tile < EDGE_TILES; tile += nWaves) {
    int base = tile * 16;
    int e = base + m0;
    int r = eidx[e];
    int c = eidx[NE + e];
    int g = batch[r];
    const float* segs[4] = { x + (size_t)r*Hdim, x + (size_t)c*Hdim,
                             eattr + (size_t)e*Hdim, u + (size_t)g*Hdim };
    // prefetch next tile's edge_attr stream (dominant HBM stream)
    if (tile + nWaves < EDGE_TILES)
      __builtin_prefetch(eattr + (size_t)(e + nWaves*16)*Hdim, 0, 1);

    // edge MLP layer 1: [16 x 256] x [256 x 64]
    Acc acc; acc_zero(acc);
    #pragma unroll
    for (int kt = 0; kt < 8; ++kt) {
      v16bf av = a_frag_segs(segs, kt, lane);
      #pragma unroll
      for (int nt = 0; nt < 4; ++nt)
        acc.c[nt] = WMMA_BF16(av, b_frag(sWE1, 256, nt, kt, lane), acc.c[nt]);
    }
    bias_gelu(acc, cbE1);
    stage_bf16(hs, acc, lane);
    gemm64_lds(acc, hs, sWE2, lane);
    bias_gelu(acc, cbE2);
    layernorm(acc, cgE, cbeE);
    store_rows(edge_out, base, acc, lane);   // edge_attr_new
    stage_bf16(hs, acc, lane);               // bf16 copy for message MLP

    // message MLP: input = [x[col] | edge_attr_new] (128)
    Acc acc2; acc_zero(acc2);
    const float* xc = x + (size_t)c * Hdim;
    #pragma unroll
    for (int kt = 0; kt < 2; ++kt) {
      v16bf av = a_frag_seg(xc, kt, lane);
      #pragma unroll
      for (int nt = 0; nt < 4; ++nt)
        acc2.c[nt] = WMMA_BF16(av, b_frag(sWM1, 128, nt, kt, lane), acc2.c[nt]);
    }
    #pragma unroll
    for (int kt = 2; kt < 4; ++kt) {
      v16bf av = a_frag_lds(hs, kt - 2, lane);
      #pragma unroll
      for (int nt = 0; nt < 4; ++nt)
        acc2.c[nt] = WMMA_BF16(av, b_frag(sWM1, 128, nt, kt, lane), acc2.c[nt]);
    }
    bias_gelu(acc2, cbM1);
    stage_bf16(hs, acc2, lane);
    gemm64_lds(acc2, hs, sWM2, lane);
    bias_gelu(acc2, cbM2);
    layernorm(acc2, cgM, cbeM);

    // scatter-add message into aggsum[row]; row index via lane shuffle
    #pragma unroll
    for (int v = 0; v < 8; ++v) {
      int rv = __shfl(r, mh + v, 32);
      #pragma unroll
      for (int nt = 0; nt < 4; ++nt)
        atomicAdd(&aggsum[(size_t)rv*Hdim + m0 + 16*nt], acc2.c[nt][v]);
      if (m0 == 0) atomicAdd(&cnt[rv], 1.0f);
    }
  }
}

// ---- node update: x_new = MLP([agg_mean | u[batch]]) ; per-graph sums ----
__global__ __launch_bounds__(128) void node_update_kernel(
    const float* __restrict__ aggsum, const float* __restrict__ cnt,
    const float* __restrict__ u, const int* __restrict__ batch,
    const __bf16* __restrict__ wAll,   // WN1|WN2 contiguous (12288 elems)
    const float* __restrict__ b1, const float* __restrict__ b2,
    const float* __restrict__ g, const float* __restrict__ be,
    float* __restrict__ x_out, float* __restrict__ gsum, float* __restrict__ gcnt)
{
  __shared__ __bf16 smem[12288 + 4*16*Hdim];   // 24KB weights + 8KB staging
  const __bf16* sW1 = smem;            // 8192 (128x64)
  const __bf16* sW2 = smem + 8192;     // 4096 ( 64x64)
  int tid  = threadIdx.x;
  int lane = tid & 31;
  int wave = tid >> 5;
  __bf16* hs = smem + 12288 + wave * 16 * Hdim;

  copy_lds(smem, wAll, 12288, tid, 128);
  __syncthreads();

  int m0 = lane & 15;
  int mh = (lane >> 4) << 3;
  float cb1[4], cb2[4], cg[4], cbe[4];
  load_coef(cb1, b1, lane); load_coef(cb2, b2, lane);
  load_coef(cg,  g,  lane); load_coef(cbe, be, lane);

  int waveId = blockIdx.x * 4 + wave;
  int nWaves = gridDim.x * 4;
  for (int tile = waveId; tile < NODE_TILES; tile += nWaves) {
    int base = tile * 16;
    int node = base + m0;
    float invc = 1.0f / fmaxf(cnt[node], 1.0f);
    int gg0 = (int)batch[node];
    const float* aseg = aggsum + (size_t)node * Hdim;
    const float* useg = u + (size_t)gg0 * Hdim;

    Acc acc; acc_zero(acc);
    #pragma unroll
    for (int kt = 0; kt < 2; ++kt) {
      v16bf av = a_frag_seg_scaled(aseg, invc, kt, lane);
      #pragma unroll
      for (int nt = 0; nt < 4; ++nt)
        acc.c[nt] = WMMA_BF16(av, b_frag(sW1, 128, nt, kt, lane), acc.c[nt]);
    }
    #pragma unroll
    for (int kt = 2; kt < 4; ++kt) {
      v16bf av = a_frag_seg(useg, kt - 2, lane);
      #pragma unroll
      for (int nt = 0; nt < 4; ++nt)
        acc.c[nt] = WMMA_BF16(av, b_frag(sW1, 128, nt, kt, lane), acc.c[nt]);
    }
    bias_gelu(acc, cb1);
    stage_bf16(hs, acc, lane);
    gemm64_lds(acc, hs, sW2, lane);
    bias_gelu(acc, cb2);
    layernorm(acc, cg, cbe);
    store_rows(x_out, base, acc, lane);

    #pragma unroll
    for (int v = 0; v < 8; ++v) {
      int gg = __shfl(gg0, mh + v, 32);
      #pragma unroll
      for (int nt = 0; nt < 4; ++nt)
        atomicAdd(&gsum[(size_t)gg*Hdim + m0 + 16*nt], acc.c[nt][v]);
      if (m0 == 0) atomicAdd(&gcnt[gg], 1.0f);
    }
  }
}

// ---- global update: u_new = MLP([u | mean_per_graph(x_new)]) ----
__global__ __launch_bounds__(128) void global_update_kernel(
    const float* __restrict__ u, const float* __restrict__ gsum,
    const float* __restrict__ gcnt,
    const __bf16* __restrict__ w1, const __bf16* __restrict__ w2,
    const float* __restrict__ b1, const float* __restrict__ b2,
    const float* __restrict__ g, const float* __restrict__ be,
    float* __restrict__ u_out)
{
  __shared__ __bf16 smem[4 * 16 * Hdim];
  int lane = threadIdx.x & 31;
  int wave = threadIdx.x >> 5;
  __bf16* hs = smem + wave * 16 * Hdim;
  int base = wave * 16;                       // 4 waves cover G=64 rows
  int m0 = lane & 15;
  int row = base + m0;
  const float* useg = u + (size_t)row * Hdim;
  float invc = 1.0f / fmaxf(gcnt[row], 1.0f);
  const float* sseg = gsum + (size_t)row * Hdim;
  float cb1[4], cb2[4], cg[4], cbe[4];
  load_coef(cb1, b1, lane); load_coef(cb2, b2, lane);
  load_coef(cg,  g,  lane); load_coef(cbe, be, lane);

  Acc acc; acc_zero(acc);
  #pragma unroll
  for (int kt = 0; kt < 2; ++kt) {
    v16bf av = a_frag_seg(useg, kt, lane);
    #pragma unroll
    for (int nt = 0; nt < 4; ++nt)
      acc.c[nt] = WMMA_BF16(av, b_frag(w1, 128, nt, kt, lane), acc.c[nt]);
  }
  #pragma unroll
  for (int kt = 2; kt < 4; ++kt) {
    v16bf av = a_frag_seg_scaled(sseg, invc, kt - 2, lane);
    #pragma unroll
    for (int nt = 0; nt < 4; ++nt)
      acc.c[nt] = WMMA_BF16(av, b_frag(w1, 128, nt, kt, lane), acc.c[nt]);
  }
  bias_gelu(acc, cb1);
  stage_bf16(hs, acc, lane);
  gemm64_lds(acc, hs, w2, lane);
  bias_gelu(acc, cb2);
  layernorm(acc, cg, cbe);
  store_rows(u_out, base, acc, lane);
}

extern "C" void kernel_launch(void* const* d_in, const int* in_sizes, int n_in,
                              void* d_out, int out_size, void* d_ws, size_t ws_size,
                              hipStream_t stream) {
  const float* x     = (const float*)d_in[0];
  const int*   eidx  = (const int*)d_in[1];
  const float* eattr = (const float*)d_in[2];
  const float* u     = (const float*)d_in[3];
  const int*   batch = (const int*)d_in[4];
  const float* eW1 = (const float*)d_in[5],  *eB1 = (const float*)d_in[6];
  const float* eW2 = (const float*)d_in[7],  *eB2 = (const float*)d_in[8];
  const float* eG  = (const float*)d_in[9],  *eBe = (const float*)d_in[10];
  const float* mW1 = (const float*)d_in[11], *mB1 = (const float*)d_in[12];
  const float* mW2 = (const float*)d_in[13], *mB2 = (const float*)d_in[14];
  const float* mG  = (const float*)d_in[15], *mBe = (const float*)d_in[16];
  const float* nW1 = (const float*)d_in[17], *nB1 = (const float*)d_in[18];
  const float* nW2 = (const float*)d_in[19], *nB2 = (const float*)d_in[20];
  const float* nG  = (const float*)d_in[21], *nBe = (const float*)d_in[22];
  const float* gW1 = (const float*)d_in[23], *gB1 = (const float*)d_in[24];
  const float* gW2 = (const float*)d_in[25], *gB2 = (const float*)d_in[26];
  const float* gG  = (const float*)d_in[27], *gBe = (const float*)d_in[28];

  char* ws = (char*)d_ws;
  __bf16* WE1 = (__bf16*)(ws + 0);        // 256x64 -> 32768 B
  __bf16* WE2 = (__bf16*)(ws + 32768);    //  64x64 ->  8192 B
  __bf16* WM1 = (__bf16*)(ws + 40960);    // 128x64 -> 16384 B
  __bf16* WM2 = (__bf16*)(ws + 57344);    //  64x64 ->  8192 B
  __bf16* WN1 = (__bf16*)(ws + 65536);    // 128x64 -> 16384 B
  __bf16* WN2 = (__bf16*)(ws + 81920);    //  64x64 ->  8192 B
  __bf16* WG1 = (__bf16*)(ws + 90112);
  __bf16* WG2 = (__bf16*)(ws + 106496);
  float* aggsum = (float*)(ws + 114688);                      // N*64 f32
  float* cnt    = (float*)(ws + 114688 + 25600000);           // N f32
  float* gsum   = (float*)(ws + 114688 + 25600000 + 400000);  // G*64 f32
  float* gcnt   = (float*)(ws + 114688 + 25600000 + 400000 + 16384); // G f32

  // zero scatter accumulators (capture-safe)
  hipMemsetAsync(aggsum, 0, 25600000 + 400000 + 16384 + 256, stream);

  // weight precompute: bf16, N-major transpose
  cvt_transpose_kernel<<<(256*Hdim + 255)/256, 256, 0, stream>>>(eW1, WE1, 256);
  cvt_transpose_kernel<<<( 64*Hdim + 255)/256, 256, 0, stream>>>(eW2, WE2, 64);
  cvt_transpose_kernel<<<(128*Hdim + 255)/256, 256, 0, stream>>>(mW1, WM1, 128);
  cvt_transpose_kernel<<<( 64*Hdim + 255)/256, 256, 0, stream>>>(mW2, WM2, 64);
  cvt_transpose_kernel<<<(128*Hdim + 255)/256, 256, 0, stream>>>(nW1, WN1, 128);
  cvt_transpose_kernel<<<( 64*Hdim + 255)/256, 256, 0, stream>>>(nW2, WN2, 64);
  cvt_transpose_kernel<<<(128*Hdim + 255)/256, 256, 0, stream>>>(gW1, WG1, 128);
  cvt_transpose_kernel<<<( 64*Hdim + 255)/256, 256, 0, stream>>>(gW2, WG2, 64);

  float* x_out    = (float*)d_out;
  float* edge_out = x_out + (size_t)NN * Hdim;
  float* u_out    = edge_out + (size_t)NE * Hdim;

  edge_node_msg_kernel<<<2048, 128, 0, stream>>>(
      x, eidx, eattr, u, batch, WE1,
      eB1, eB2, eG, eBe, mB1, mB2, mG, mBe,
      edge_out, aggsum, cnt);

  node_update_kernel<<<512, 128, 0, stream>>>(
      aggsum, cnt, u, batch, WN1, nB1, nB2, nG, nBe,
      x_out, gsum, gcnt);

  global_update_kernel<<<1, 128, 0, stream>>>(
      u, gsum, gcnt, WG1, WG2, gB1, gB2, gG, gBe, u_out);
}
